// ResidualVectorQuantizer_11123965297179
// MI455X (gfx1250) — matrix operand order repaired and checked by reading, not compiled, not run on hardware
//
#include <hip/hip_runtime.h>

#define NROWS 65536
#define EDIM  512
#define NE    1024
#define NUMQ  4
#define MU    0.25f

#define AS1 __attribute__((address_space(1)))
#define AS3 __attribute__((address_space(3)))

typedef __attribute__((ext_vector_type(16))) __bf16 v16bf;
typedef __attribute__((ext_vector_type(8)))  float  v8f;
typedef __attribute__((ext_vector_type(4)))  int    v4i;

union FragU { v16bf f; uint4 u[2]; };

__device__ __forceinline__ unsigned short f2bf(float x) {
  unsigned int u = __builtin_bit_cast(unsigned int, x);
  unsigned int r = u + 0x7FFFu + ((u >> 16) & 1u);   // round-to-nearest-even
  return (unsigned short)(r >> 16);
}

// A operand: 16x32 (MxK) bf16 tile. lane<16 -> row M=lane, K = {0..7,16..23};
// lane>=16 -> same row, K = {8..15,24..31}.  (ISA 7.12.2 16-bit A layout)
__device__ __forceinline__ v16bf load_frag_a(const unsigned short* base, int row0, int lane) {
  int r  = row0 + (lane & 15);
  int kb = (lane >> 4) ? 8 : 0;
  const unsigned short* p = base + r * 32 + kb;
  FragU f;
  f.u[0] = *(const uint4*)(p);        // K kb..kb+7
  f.u[1] = *(const uint4*)(p + 16);   // K kb+16..kb+23
  return f.f;
}

// B operand: 32x16 (KxN) bf16; tile stored row-major as [code][32 K].
// lane<16 -> column N=lane with K = {0..7,16..23}; lane>=16 -> K = {8..15,24..31}.
__device__ __forceinline__ v16bf load_frag_b(const unsigned short* base, int row0, int lane) {
  int r  = row0 + (lane & 15);
  int kb = (lane >> 4) ? 8 : 0;
  const unsigned short* p = base + r * 32 + kb;
  FragU f;
  f.u[0] = *(const uint4*)(p);
  f.u[1] = *(const uint4*)(p + 16);
  return f.f;
}

// async global->LDS stage of one 128x32 bf16 tile (512 b128 units, 2/thread)
__device__ __forceinline__ void stage_tile_async(unsigned short* lds,
                                                 const unsigned short* g, int tid) {
#pragma unroll
  for (int r = 0; r < 2; ++r) {
    int u = tid + r * 256;
    int row = u >> 2, col = (u & 3) * 8;
    __builtin_amdgcn_global_load_async_to_lds_b128(
        (AS1 v4i*)(unsigned short*)(g + (size_t)row * EDIM + col),
        (AS3 v4i*)(lds + row * 32 + col), 0, 0);
  }
}

__device__ __forceinline__ void wait_async0() {
  asm volatile("s_wait_asynccnt 0x0" ::: "memory");
}

// ---------------- prep kernels ----------------

__global__ void rvq_init(float* __restrict__ loss_acc) {
  if (threadIdx.x < NUMQ) loss_acc[threadIdx.x] = 0.0f;
}

// one wave per codebook row: f32->bf16 convert + ||e||^2
__global__ __launch_bounds__(256) void rvq_prep_cb(const float* __restrict__ cb,
                                                   unsigned short* __restrict__ cbb,
                                                   float* __restrict__ en2) {
  int wv = threadIdx.x >> 5, lane = threadIdx.x & 31;
  int row = blockIdx.x * 8 + wv;                 // 0..NUMQ*NE-1
  const float* src = cb + (size_t)row * EDIM;
  unsigned short* dst = cbb + (size_t)row * EDIM;
  float s = 0.0f;
#pragma unroll
  for (int c4 = 0; c4 < 4; ++c4) {
    int i = (c4 * 32 + lane) * 4;
    float4 v = *(const float4*)(src + i);
    s += v.x * v.x + v.y * v.y + v.z * v.z + v.w * v.w;
    uint2 p;
    p.x = (unsigned)f2bf(v.x) | ((unsigned)f2bf(v.y) << 16);
    p.y = (unsigned)f2bf(v.z) | ((unsigned)f2bf(v.w) << 16);
    *(uint2*)(dst + i) = p;
  }
#pragma unroll
  for (int off = 16; off; off >>= 1) s += __shfl_xor(s, off, 32);
  if (lane == 0) en2[row] = s;
}

// one wave per data row: ||x||^2 and bf16 copy of x (stage-0 A operand)
__global__ __launch_bounds__(256) void rvq_prep_x(const float* __restrict__ x,
                                                  unsigned short* __restrict__ rbf,
                                                  float* __restrict__ rn2) {
  int wv = threadIdx.x >> 5, lane = threadIdx.x & 31;
  int n = blockIdx.x * 8 + wv;
  const float* src = x + (size_t)n * EDIM;
  unsigned short* dst = rbf + (size_t)n * EDIM;
  float s = 0.0f;
#pragma unroll
  for (int c4 = 0; c4 < 4; ++c4) {
    int i = (c4 * 32 + lane) * 4;
    float4 v = *(const float4*)(src + i);
    s += v.x * v.x + v.y * v.y + v.z * v.z + v.w * v.w;
    uint2 p;
    p.x = (unsigned)f2bf(v.x) | ((unsigned)f2bf(v.y) << 16);
    p.y = (unsigned)f2bf(v.z) | ((unsigned)f2bf(v.w) << 16);
    *(uint2*)(dst + i) = p;
  }
#pragma unroll
  for (int off = 16; off; off >>= 1) s += __shfl_xor(s, off, 32);
  if (lane == 0) rn2[n] = s;
}

// ---------------- distance GEMM (bf16 WMMA, f32 accumulate) ----------------
// block tile 128(M) x 128(N), K-step 32; 8 waves as 2(M) x 4(N), wave tile 64x32.
// Both operands bf16 in global; async-copied into double-buffered LDS.

__global__ __launch_bounds__(256) void rvq_gemm(const unsigned short* __restrict__ abf,
                                                const unsigned short* __restrict__ cbb,
                                                const float* __restrict__ rn2,
                                                const float* __restrict__ en2,
                                                float* __restrict__ dist, int q) {
  __shared__ __align__(16) unsigned short lA[2][128 * 32];
  __shared__ __align__(16) unsigned short lB[2][128 * 32];

  const int bm = blockIdx.x * 128;
  const int bn = blockIdx.y * 128;
  const int tid = threadIdx.x, lane = tid & 31, wv = tid >> 5;
  const int wm = (wv & 1) * 64;    // wave M offset inside block tile
  const int wn = (wv >> 1) * 32;   // wave N offset inside block tile

  const unsigned short* aG = abf + (size_t)bm * EDIM;
  const unsigned short* bG = cbb + (size_t)bn * EDIM;

  v8f c[4][2] = {};                // 4 M-frags x 2 N-frags of 16x16 f32

  // prologue: async-stage K-step 0 into buffer 0
  stage_tile_async(lA[0], aG, tid);
  stage_tile_async(lB[0], bG, tid);

  for (int kt = 0; kt < EDIM / 32; ++kt) {
    wait_async0();                 // own async copies into buf `cur` complete
    __syncthreads();               // everyone's copies complete
    const int cur = kt & 1;
    if (kt + 1 < EDIM / 32) {      // prefetch next K-step into other buffer
      stage_tile_async(lA[cur ^ 1], aG + (kt + 1) * 32, tid);
      stage_tile_async(lB[cur ^ 1], bG + (kt + 1) * 32, tid);
    }

    v16bf a[4], b[2];
#pragma unroll
    for (int mf = 0; mf < 4; ++mf) a[mf] = load_frag_a(lA[cur], wm + mf * 16, lane);
#pragma unroll
    for (int nf = 0; nf < 2; ++nf) b[nf] = load_frag_b(lB[cur], wn + nf * 16, lane);

#pragma unroll
    for (int mf = 0; mf < 4; ++mf)
#pragma unroll
      for (int nf = 0; nf < 2; ++nf)
        c[mf][nf] = __builtin_amdgcn_wmma_f32_16x16x32_bf16(
            false, a[mf], false, b[nf], (short)0, c[mf][nf], false, false);
    // no end-of-loop barrier needed: next iteration's barrier is only passable
    // after every wave consumed its ds_loads (s_wait_dscnt before the WMMAs),
    // and prefetch writes always target the other buffer.
  }

  // epilogue: d = ||r||^2 + ||e||^2 - 2*dot; C layout: row=j+8*(lane>=16), col=lane&15
  const int h = lane >> 4, ln = lane & 15;
#pragma unroll
  for (int nf = 0; nf < 2; ++nf) {
    int col = bn + wn + nf * 16 + ln;
    float e2 = en2[col];
#pragma unroll
    for (int mf = 0; mf < 4; ++mf) {
#pragma unroll
      for (int j = 0; j < 8; ++j) {
        int row = bm + wm + mf * 16 + j + 8 * h;
        float d = rn2[row] + e2 - 2.0f * c[mf][nf][j];
        dist[((size_t)row * NUMQ + q) * NE + col] = d;
      }
    }
  }
}

// ---------------- argmin + gather + residual/xq/loss update ----------------
// one wave32 per row

__global__ __launch_bounds__(256) void rvq_update(const float* __restrict__ resid_in,
                                                  const float* __restrict__ cb,
                                                  const float* __restrict__ dist, int q,
                                                  float* __restrict__ resid_out,
                                                  unsigned short* __restrict__ rbf,
                                                  float* __restrict__ out_xq,
                                                  int* __restrict__ out_idx,
                                                  float* __restrict__ rn2,
                                                  float* __restrict__ loss_acc) {
  int wv = threadIdx.x >> 5, lane = threadIdx.x & 31;
  int n = blockIdx.x * 8 + wv;
  const float* drow = dist + ((size_t)n * NUMQ + q) * NE;

  float bd = 3.402823466e38f;
  int bk = 0;
#pragma unroll
  for (int t = 0; t < NE / 32; ++t) {        // lane scans k = lane, lane+32, ... (ascending)
    int k = t * 32 + lane;
    float d = drow[k];
    if (d < bd) { bd = d; bk = k; }
  }
#pragma unroll
  for (int off = 16; off; off >>= 1) {       // first-min tie-break (matches jnp.argmin)
    float od = __shfl_xor(bd, off, 32);
    int   ok = __shfl_xor(bk, off, 32);
    if (od < bd || (od == bd && ok < bk)) { bd = od; bk = ok; }
  }

  const float* e  = cb + ((size_t)q * NE + bk) * EDIM;   // gather codeword (f32)
  const float* r  = resid_in  + (size_t)n * EDIM;
  float*       ro = resid_out + (size_t)n * EDIM;
  unsigned short* rb = rbf    + (size_t)n * EDIM;
  float*       xo = out_xq    + (size_t)n * EDIM;

  float s = 0.0f;                            // sum (r - xq)^2 == row SSE == next ||r||^2
#pragma unroll
  for (int c4 = 0; c4 < 4; ++c4) {
    int i = (c4 * 32 + lane) * 4;
    float4 xv = *(const float4*)(e + i);
    float4 rv = *(const float4*)(r + i);
    float4 nr;
    nr.x = rv.x - xv.x; nr.y = rv.y - xv.y; nr.z = rv.z - xv.z; nr.w = rv.w - xv.w;
    s += nr.x * nr.x + nr.y * nr.y + nr.z * nr.z + nr.w * nr.w;
    *(float4*)(ro + i) = nr;
    uint2 p;                                 // bf16 residual for next stage's A operand
    p.x = (unsigned)f2bf(nr.x) | ((unsigned)f2bf(nr.y) << 16);
    p.y = (unsigned)f2bf(nr.z) | ((unsigned)f2bf(nr.w) << 16);
    *(uint2*)(rb + i) = p;
    if (q == 0) {
      *(float4*)(xo + i) = xv;
    } else {
      float4 a = *(const float4*)(xo + i);
      a.x += xv.x; a.y += xv.y; a.z += xv.z; a.w += xv.w;
      *(float4*)(xo + i) = a;
    }
  }
#pragma unroll
  for (int off = 16; off; off >>= 1) s += __shfl_xor(s, off, 32);
  if (lane == 0) {
    rn2[n] = s;                              // ||residual||^2 for next stage
    atomicAdd(loss_acc + q, s);
    out_idx[(size_t)n * NUMQ + q] = bk;
  }
}

__global__ void rvq_finalize(const float* __restrict__ loss_acc, float* __restrict__ out_loss) {
  if (threadIdx.x == 0) {
    float t = 0.0f;
    for (int q = 0; q < NUMQ; ++q)
      t += (1.0f + MU) * loss_acc[q] / ((float)NROWS * (float)EDIM);
    *out_loss = t / (float)NUMQ;
  }
}

// ---------------- launch ----------------

extern "C" void kernel_launch(void* const* d_in, const int* in_sizes, int n_in,
                              void* d_out, int out_size, void* d_ws, size_t ws_size,
                              hipStream_t stream) {
  (void)in_sizes; (void)n_in; (void)out_size; (void)ws_size;
  const float* x  = (const float*)d_in[0];     // [65536, 512]
  const float* cb = (const float*)d_in[1];     // [4, 1024, 512]

  // workspace carve-out (~206 MB)
  char* w = (char*)d_ws;
  float* resid = (float*)w;                 w += (size_t)NROWS * EDIM * sizeof(float);
  unsigned short* rbf = (unsigned short*)w; w += (size_t)NROWS * EDIM * sizeof(unsigned short);
  unsigned short* cbb = (unsigned short*)w; w += (size_t)NUMQ * NE * EDIM * sizeof(unsigned short);
  float* rn2 = (float*)w;                   w += (size_t)NROWS * sizeof(float);
  float* en2 = (float*)w;                   w += (size_t)NUMQ * NE * sizeof(float);
  float* loss_acc = (float*)w;              w += (size_t)NUMQ * sizeof(float);

  // output layout: x_q | mean_loss | all_indices | all_distances
  float* out_xq   = (float*)d_out;
  float* out_loss = out_xq + (size_t)NROWS * EDIM;
  int*   out_idx  = (int*)(out_loss + 1);
  float* out_dist = (float*)(out_idx + (size_t)NROWS * NUMQ);

  rvq_init<<<1, 32, 0, stream>>>(loss_acc);
  rvq_prep_cb<<<NUMQ * NE / 8, 256, 0, stream>>>(cb, cbb, en2);
  rvq_prep_x<<<NROWS / 8, 256, 0, stream>>>(x, rbf, rn2);

  for (int q = 0; q < NUMQ; ++q) {
    const float* rin = (q == 0) ? x : resid;
    rvq_gemm<<<dim3(NROWS / 128, NE / 128), 256, 0, stream>>>(
        rbf, cbb + (size_t)q * NE * EDIM, rn2, en2 + q * NE, out_dist, q);
    rvq_update<<<NROWS / 8, 256, 0, stream>>>(
        rin, cb, out_dist, q, resid, rbf, out_xq, out_idx, rn2, loss_acc);
  }
  rvq_finalize<<<1, 32, 0, stream>>>(loss_acc, out_loss);
}
